// STEMAttention_50663434223715
// MI455X (gfx1250) — matrix operand order repaired and checked by reading, not compiled, not run on hardware
//
#include <hip/hip_runtime.h>

// ---------------------------------------------------------------------------
// Attention block (S=4096, E=960, H=15, KVH=5, D=64, GQA groups=3) for
// MI455X / gfx1250.  All matmuls: v_wmma_f32_16x16x32_bf16.
//
// Round-2 change: all fp32->bf16 conversion is hoisted OUT of the WMMA hot
// loops.  Inputs are pre-converted once; every intermediate (Q/K post-RoPE,
// V transposed, attn) is stored bf16 by its producer.  Hot-loop fragment
// loads are raw global_load_b128 into operand VGPRs (no VALU), halving both
// the conversion work and the streamed bytes.
//
// Workspace (bf16 elements):
//   Xb    [4096][960]      hidden in bf16
//   Wqkvb [1600][960]      concat(Wq,Wk,Wv) in bf16
//   Wob   [960][960]       Wo in bf16
//   Qb    [15][4096][64]   RoPE'd queries
//   Kb    [ 5][4096][64]   RoPE'd keys
//   Vt    [ 5][64][4096]   values transposed (contiguous PV B-fragments)
//   attnb [4096][960]      attention output
// Total ~16.9M bf16 = ~34 MiB of d_ws.
// ---------------------------------------------------------------------------

typedef __attribute__((ext_vector_type(16))) __bf16 v16bf;
typedef __attribute__((ext_vector_type(8)))  float  v8f;

#define WMMA_BF16(a, b, c) \
  __builtin_amdgcn_wmma_f32_16x16x32_bf16(false, (a), false, (b), (short)0, (c), false, false)

union BF16Frag  { v16bf v; uint4 u[2]; };
union BF16Pack8 { __bf16 b[8]; uint4 u; };
union BF16Pack4 { __bf16 b[4]; uint2 u; };

// A fragment (16-bit 16x32): per-lane K chunks at kofs..+7 and kofs+16..+23
// (kofs already includes (lane>>4)*8).  Two b128 loads, no VALU.
__device__ __forceinline__ v16bf fragA(const __bf16* rowptr, int kofs) {
  BF16Frag f;
  f.u[0] = *reinterpret_cast<const uint4*>(rowptr + kofs);
  f.u[1] = *reinterpret_cast<const uint4*>(rowptr + kofs + 16);
  return f.v;
}

// B fragment (16-bit 32x16): K chunks at p..+7 and p+16..+23 (p includes (lane>>4)*8).
__device__ __forceinline__ v16bf fragB(const __bf16* p) {
  BF16Frag f;
  f.u[0] = *reinterpret_cast<const uint4*>(p);
  f.u[1] = *reinterpret_cast<const uint4*>(p + 16);
  return f.v;
}

// ---------------------------------------------------------------------------
// Kernel 0: fp32 -> bf16 pre-conversion (elementwise, 4 per thread).
// ---------------------------------------------------------------------------
__global__ __launch_bounds__(256) void cvt_bf16_kernel(
    const float* __restrict__ src, __bf16* __restrict__ dst, int n)
{
  const int i = (blockIdx.x * 256 + threadIdx.x) * 4;
  if (i < n) {
    const float4 x = *reinterpret_cast<const float4*>(src + i);
    BF16Pack4 o;
    o.b[0] = (__bf16)x.x; o.b[1] = (__bf16)x.y;
    o.b[2] = (__bf16)x.z; o.b[3] = (__bf16)x.w;
    *reinterpret_cast<uint2*>(dst + i) = o.u;
  }
}

// ---------------------------------------------------------------------------
// Kernel 1: fused QKV projection + RoPE (+ V transpose).
// grid (16, 25), block (32, 8).  Wave = 32 rows x 64 features (one head tile).
// N-tiles 0..14 -> Q heads, 15..19 -> K heads, 20..24 -> V heads.
// ---------------------------------------------------------------------------
__global__ __launch_bounds__(256) void qkv_rope_kernel(
    const __bf16* __restrict__ Xb, const __bf16* __restrict__ Wqkvb,
    __bf16* __restrict__ Qb, __bf16* __restrict__ Kb, __bf16* __restrict__ Vt)
{
  const int lane  = threadIdx.x & 31;
  const int r     = lane & 15;
  const int hf    = lane >> 4;
  const int mbase = blockIdx.x * 256 + threadIdx.y * 32;
  const int ntile = blockIdx.y;          // 0..24
  const int nbase = ntile * 64;

  const __bf16* wrow[4];
#pragma unroll
  for (int j = 0; j < 4; ++j) wrow[j] = Wqkvb + (size_t)(nbase + j * 16 + r) * 960;
  const __bf16* arow0 = Xb + (size_t)(mbase + r) * 960;
  const __bf16* arow1 = Xb + (size_t)(mbase + 16 + r) * 960;

  v8f acc[2][4];
#pragma unroll
  for (int mi = 0; mi < 2; ++mi)
#pragma unroll
    for (int j = 0; j < 4; ++j)
#pragma unroll
      for (int i = 0; i < 8; ++i) acc[mi][j][i] = 0.0f;

  for (int kk = 0; kk < 960; kk += 32) {
    const v16bf a0 = fragA(arow0, kk + hf * 8);
    const v16bf a1 = fragA(arow1, kk + hf * 8);
    v16bf b[4];
#pragma unroll
    for (int j = 0; j < 4; ++j) b[j] = fragB(wrow[j] + kk + hf * 8);
#pragma unroll
    for (int j = 0; j < 4; ++j) {
      acc[0][j] = WMMA_BF16(a0, b[j], acc[0][j]);
      acc[1][j] = WMMA_BF16(a1, b[j], acc[1][j]);
    }
  }

  if (ntile < 20) {
    // RoPE on Q / K head tile (fp32), store bf16.  d = j*16 + r.
    const bool isQ = ntile < 15;
    __bf16* base = isQ ? (Qb + (size_t)ntile * 4096 * 64)
                       : (Kb + (size_t)(ntile - 15) * 4096 * 64);
    const float inv0 = powf(10000.0f, -(float)r        * (1.0f / 32.0f));
    const float inv1 = powf(10000.0f, -(float)(r + 16) * (1.0f / 32.0f));
#pragma unroll
    for (int mi = 0; mi < 2; ++mi) {
#pragma unroll
      for (int v = 0; v < 8; ++v) {
        const int s = mbase + mi * 16 + hf * 8 + v;
        float sn0, cs0, sn1, cs1;
        sincosf((float)s * inv0, &sn0, &cs0);
        sincosf((float)s * inv1, &sn1, &cs1);
        const float x0 = acc[mi][0][v], x1 = acc[mi][1][v];
        const float x2 = acc[mi][2][v], x3 = acc[mi][3][v];
        __bf16* row = base + (size_t)s * 64;
        row[ 0 + r] = (__bf16)(x0 * cs0 - x2 * sn0);   // d = r       (< 32)
        row[16 + r] = (__bf16)(x1 * cs1 - x3 * sn1);   // d = 16 + r  (< 32)
        row[32 + r] = (__bf16)(x2 * cs0 + x0 * sn0);   // d = 32 + r
        row[48 + r] = (__bf16)(x3 * cs1 + x1 * sn1);   // d = 48 + r
      }
    }
  } else {
    // V head: store transposed [kvh][d][s]; 8 consecutive s per (j,lane) -> b128.
    __bf16* base = Vt + (size_t)(ntile - 20) * 64 * 4096;
#pragma unroll
    for (int mi = 0; mi < 2; ++mi) {
#pragma unroll
      for (int j = 0; j < 4; ++j) {
        const int d = j * 16 + r;
        BF16Pack8 o;
#pragma unroll
        for (int v = 0; v < 8; ++v) o.b[v] = (__bf16)acc[mi][j][v];
        __bf16* col = base + (size_t)d * 4096 + mbase + mi * 16 + hf * 8;
        *reinterpret_cast<uint4*>(col) = o.u;
      }
    }
  }
}

// ---------------------------------------------------------------------------
// Kernel 2: causal flash attention.  grid (256, 15), block (32).
// One wave: 16 queries of head h, online softmax over 32-key tiles.
// ---------------------------------------------------------------------------
__global__ __launch_bounds__(32) void attn_kernel(
    const __bf16* __restrict__ Qb, const __bf16* __restrict__ Kb,
    const __bf16* __restrict__ Vt, __bf16* __restrict__ attnb)
{
  __shared__ __bf16 P[16][32];         // probs staging: C-layout -> A-layout
  const int lane  = threadIdx.x & 31;
  const int r     = lane & 15;
  const int hf    = lane >> 4;
  const int h     = blockIdx.y;
  const int kvh   = h / 3;             // GQA: 3 query heads per kv head
  const int qbase = blockIdx.x * 16;

  const __bf16* qrow = Qb + ((size_t)h * 4096 + qbase + r) * 64;
  const v16bf aq0 = fragA(qrow,      hf * 8);
  const v16bf aq1 = fragA(qrow, 32 + hf * 8);

  const __bf16* Kh = Kb + (size_t)kvh * 4096 * 64;
  const __bf16* Vh = Vt + (size_t)kvh * 64 * 4096;

  float m_i[8], l_i[8];
  v8f o[4];
#pragma unroll
  for (int v = 0; v < 8; ++v) { m_i[v] = -1e30f; l_i[v] = 0.0f; }
#pragma unroll
  for (int dj = 0; dj < 4; ++dj)
#pragma unroll
    for (int i = 0; i < 8; ++i) o[dj][i] = 0.0f;

  const int nkt = (qbase + 47) >> 5;   // key tiles covering keys <= qbase+15
  for (int kt = 0; kt < nkt; ++kt) {
    const int kb = kt * 32;

    v8f sc[2];
#pragma unroll
    for (int nj = 0; nj < 2; ++nj)
#pragma unroll
      for (int i = 0; i < 8; ++i) sc[nj][i] = 0.0f;
#pragma unroll
    for (int nj = 0; nj < 2; ++nj) {
      const __bf16* krow = Kh + (size_t)(kb + nj * 16 + r) * 64;
      const v16bf b0 = fragB(krow + hf * 8);         // d = 0..31 slice
      const v16bf b1 = fragB(krow + 32 + hf * 8);    // d = 32..63 slice
      sc[nj] = WMMA_BF16(aq0, b0, sc[nj]);
      sc[nj] = WMMA_BF16(aq1, b1, sc[nj]);
    }

    float scale[8];
#pragma unroll
    for (int v = 0; v < 8; ++v) {
      const int q = qbase + hf * 8 + v;              // C-layout row
      float s0 = sc[0][v] * 0.125f;                  // 1/sqrt(64)
      float s1 = sc[1][v] * 0.125f;
      if (kb + r      > q) s0 = -1e9f;
      if (kb + 16 + r > q) s1 = -1e9f;
      float mt = fmaxf(s0, s1);
      mt = fmaxf(mt, __shfl_xor(mt, 1));
      mt = fmaxf(mt, __shfl_xor(mt, 2));
      mt = fmaxf(mt, __shfl_xor(mt, 4));
      mt = fmaxf(mt, __shfl_xor(mt, 8));             // reduce within 16-lane half
      const float mn = fmaxf(m_i[v], mt);
      const float sf = __expf(m_i[v] - mn);
      m_i[v] = mn;
      const float p0 = __expf(s0 - mn);
      const float p1 = __expf(s1 - mn);
      float rs = p0 + p1;
      rs += __shfl_xor(rs, 1);
      rs += __shfl_xor(rs, 2);
      rs += __shfl_xor(rs, 4);
      rs += __shfl_xor(rs, 8);
      l_i[v] = l_i[v] * sf + rs;
      scale[v] = sf;
      P[hf * 8 + v][r]      = (__bf16)p0;
      P[hf * 8 + v][16 + r] = (__bf16)p1;
    }
#pragma unroll
    for (int dj = 0; dj < 4; ++dj)
#pragma unroll
      for (int v = 0; v < 8; ++v) o[dj][v] *= scale[v];

    __syncthreads();
    BF16Frag ap;                        // A-layout read of P (2x ds_load_b128)
    ap.u[0] = *reinterpret_cast<const uint4*>(&P[r][hf * 8]);
    ap.u[1] = *reinterpret_cast<const uint4*>(&P[r][16 + hf * 8]);
    __syncthreads();

#pragma unroll
    for (int dj = 0; dj < 4; ++dj) {
      const __bf16* vrow = Vh + (size_t)(dj * 16 + r) * 4096 + kb + hf * 8;
      o[dj] = WMMA_BF16(ap.v, fragB(vrow), o[dj]);
    }
  }

#pragma unroll
  for (int v = 0; v < 8; ++v) {
    const float inv = 1.0f / l_i[v];
    const int s = qbase + hf * 8 + v;
    __bf16* dst = attnb + (size_t)s * 960 + h * 64;
#pragma unroll
    for (int dj = 0; dj < 4; ++dj) dst[dj * 16 + r] = (__bf16)(o[dj][v] * inv);
  }
}

// ---------------------------------------------------------------------------
// Kernel 3: output projection attn[4096x960] @ Wo^T -> fp32 out.
// grid (16, 15), block (32, 8).
// ---------------------------------------------------------------------------
__global__ __launch_bounds__(256) void oproj_kernel(
    const __bf16* __restrict__ A, const __bf16* __restrict__ Wob,
    float* __restrict__ O)
{
  const int lane  = threadIdx.x & 31;
  const int r     = lane & 15;
  const int hf    = lane >> 4;
  const int mbase = blockIdx.x * 256 + threadIdx.y * 32;
  const int nbase = blockIdx.y * 64;

  const __bf16* arow0 = A + (size_t)(mbase + r) * 960;
  const __bf16* arow1 = A + (size_t)(mbase + 16 + r) * 960;
  const __bf16* wrow[4];
#pragma unroll
  for (int j = 0; j < 4; ++j) wrow[j] = Wob + (size_t)(nbase + j * 16 + r) * 960;

  v8f acc[2][4];
#pragma unroll
  for (int mi = 0; mi < 2; ++mi)
#pragma unroll
    for (int j = 0; j < 4; ++j)
#pragma unroll
      for (int i = 0; i < 8; ++i) acc[mi][j][i] = 0.0f;

  for (int kk = 0; kk < 960; kk += 32) {
    const v16bf a0 = fragA(arow0, kk + hf * 8);
    const v16bf a1 = fragA(arow1, kk + hf * 8);
#pragma unroll
    for (int j = 0; j < 4; ++j) {
      const v16bf b = fragB(wrow[j] + kk + hf * 8);
      acc[0][j] = WMMA_BF16(a0, b, acc[0][j]);
      acc[1][j] = WMMA_BF16(a1, b, acc[1][j]);
    }
  }

#pragma unroll
  for (int mi = 0; mi < 2; ++mi)
#pragma unroll
    for (int v = 0; v < 8; ++v) {
      const int s = mbase + mi * 16 + hf * 8 + v;
      float* row = O + (size_t)s * 960 + nbase;
#pragma unroll
      for (int j = 0; j < 4; ++j) row[j * 16 + r] = acc[mi][j][v];
    }
}

// ---------------------------------------------------------------------------
extern "C" void kernel_launch(void* const* d_in, const int* in_sizes, int n_in,
                              void* d_out, int out_size, void* d_ws, size_t ws_size,
                              hipStream_t stream) {
  (void)in_sizes; (void)n_in; (void)out_size; (void)ws_size;
  const float* hidden = (const float*)d_in[0];   // [4096, 960]
  const float* Wq     = (const float*)d_in[1];   // [960, 960]
  const float* Wk     = (const float*)d_in[2];   // [320, 960]
  const float* Wv     = (const float*)d_in[3];   // [320, 960]
  const float* Wo     = (const float*)d_in[4];   // [960, 960]
  // d_in[5] (attention_mask) and d_in[6] (position_ids) are recomputed on-chip.

  __bf16* p     = (__bf16*)d_ws;
  __bf16* Xb    = p;  p += (size_t)4096 * 960;        // 3,932,160
  __bf16* Wqkvb = p;  p += (size_t)1600 * 960;        // 1,536,000
  __bf16* Wob   = p;  p += (size_t)960 * 960;         //   921,600
  __bf16* Qb    = p;  p += (size_t)15 * 4096 * 64;    // 3,932,160
  __bf16* Kbuf  = p;  p += (size_t)5 * 4096 * 64;     // 1,310,720
  __bf16* Vt    = p;  p += (size_t)5 * 64 * 4096;     // 1,310,720
  __bf16* attnb = p;                                  // 3,932,160

  const int nX = 4096 * 960, nWq = 960 * 960, nWkv = 320 * 960, nWo = 960 * 960;
  cvt_bf16_kernel<<<nX   / 1024, 256, 0, stream>>>(hidden, Xb, nX);
  cvt_bf16_kernel<<<nWq  / 1024, 256, 0, stream>>>(Wq, Wqkvb, nWq);
  cvt_bf16_kernel<<<nWkv / 1024, 256, 0, stream>>>(Wk, Wqkvb + (size_t)960 * 960, nWkv);
  cvt_bf16_kernel<<<nWkv / 1024, 256, 0, stream>>>(Wv, Wqkvb + (size_t)1280 * 960, nWkv);
  cvt_bf16_kernel<<<nWo  / 1024, 256, 0, stream>>>(Wo, Wob, nWo);

  qkv_rope_kernel<<<dim3(16, 25), dim3(32, 8), 0, stream>>>(Xb, Wqkvb, Qb, Kbuf, Vt);
  attn_kernel<<<dim3(256, 15), dim3(32), 0, stream>>>(Qb, Kbuf, Vt, attnb);
  oproj_kernel<<<dim3(16, 15), dim3(32, 8), 0, stream>>>(attnb, Wob, (float*)d_out);
}
